// AttentionNetLayer_20959440404501
// MI455X (gfx1250) — compile-verified, matrix operand order checked by reading, not run on hardware
//
#include <hip/hip_runtime.h>
#include <hip/hip_bf16.h>

typedef _Float16 f16;
typedef __attribute__((ext_vector_type(16))) _Float16 v16h;
typedef __attribute__((ext_vector_type(8)))  _Float16 v8h;
typedef __attribute__((ext_vector_type(8)))  float    v8f;

#define Bdim 4
#define Npts 8192
#define Cch  16
#define G    1024
#define S    32
#define NH   4
#define DMODEL 128
#define MROWS  (Bdim*G*S)      // 131072
#define MT     4               // m-tiles per wave in GEMM
#define RSCALE 0.08838834764831845f   // 1/sqrt(128)

// ---------------------------------------------------------------------------
// Weight prep: rearrange W [K,128] f32 into WMMA B-fragment layout (f16).
// frag[((ntile*KC + kc)*32 + lane)*16 + j] = W[kc*32 + kmap(lane,j)][ntile*16 + (lane&15)]
// kmap(lane,j) = ((j&8)<<1) + ((lane&16)>>1) + (j&7)   (16-bit B layout, wave32)
// ---------------------------------------------------------------------------
__global__ void prep_frag(const float* __restrict__ W, int K, int KC, f16* __restrict__ frag) {
    int tid = blockIdx.x * blockDim.x + threadIdx.x;
    int total = 8 * KC * 512;               // ntiles(8) * KC * 32 lanes * 16 elems
    if (tid >= total) return;
    int ntile = tid / (KC * 512);
    int rem   = tid - ntile * KC * 512;
    int kc    = rem >> 9;
    int l     = (rem >> 4) & 31;
    int j     = rem & 15;
    int k = kc * 32 + ((j & 8) << 1) + ((l & 16) >> 1) + (j & 7);
    int n = ntile * 16 + (l & 15);
    float v = (k < K) ? W[k * DMODEL + n] : 0.f;
    frag[tid] = (f16)v;
}

// ---------------------------------------------------------------------------
// FPS: one block per batch, dist[] resident in LDS, 1024 serial argmax steps.
// Matches reference scan: emit far, min-update dist, far = argmax(dist).
// ---------------------------------------------------------------------------
__global__ __launch_bounds__(256) void fps_kernel(const float* __restrict__ xyz,
                                                  float* __restrict__ nx_ws,
                                                  float* __restrict__ nx_out) {
    __shared__ float dist[Npts];
    __shared__ float rv[256];
    __shared__ int   ri[256];
    __shared__ int   sfar;
    int b = blockIdx.x, t = threadIdx.x;
    const float* X = xyz + (size_t)b * Npts * 3;
    for (int j = t; j < Npts; j += 256) dist[j] = 1e10f;
    if (t == 0) sfar = 0;
    __syncthreads();
    for (int i = 0; i < G; ++i) {
        int far = sfar;
        float cx = X[far*3+0], cy = X[far*3+1], cz = X[far*3+2];
        if (t < 3) {
            float v = X[far*3 + t];
            nx_ws [(b*G + i)*3 + t] = v;
            nx_out[(b*G + i)*3 + t] = v;
        }
        float bv = -1.f; int bi = 0x7fffffff;
        for (int j = t; j < Npts; j += 256) {
            float dx = X[j*3+0]-cx, dy = X[j*3+1]-cy, dz = X[j*3+2]-cz;
            float d  = dx*dx + dy*dy + dz*dz;
            float dd = dist[j]; dd = dd < d ? dd : d; dist[j] = dd;
            if (dd > bv || (dd == bv && j < bi)) { bv = dd; bi = j; }
        }
        rv[t] = bv; ri[t] = bi;
        __syncthreads();
        for (int s = 128; s > 0; s >>= 1) {
            if (t < s) {
                float v2 = rv[t+s]; int i2 = ri[t+s];
                if (v2 > rv[t] || (v2 == rv[t] && i2 < ri[t])) { rv[t] = v2; ri[t] = i2; }
            }
            __syncthreads();
        }
        if (t == 0) sfar = ri[0];
        __syncthreads();
    }
}

// ---------------------------------------------------------------------------
// Ball query + group: one wave32 per center. Ordered ballot compaction gives
// the first 32 in-radius indices in ascending order (== reference top_k trick).
// Builds padded f16 x0 [MROWS, 32] (19 real features + zero pad).
// ---------------------------------------------------------------------------
__global__ __launch_bounds__(32) void group_kernel(const float* __restrict__ xyz,
                                                   const float* __restrict__ points,
                                                   const float* __restrict__ nx,
                                                   f16* __restrict__ x0,
                                                   float* __restrict__ idx_out) {
    __shared__ int sidx[32];
    int c = blockIdx.x;            // b*G + g
    int b = c >> 10;
    int lane = threadIdx.x;
    const float* X = xyz + (size_t)b * Npts * 3;
    float cx = nx[c*3+0], cy = nx[c*3+1], cz = nx[c*3+2];
    int count = 0;
    for (int base = 0; base < Npts; base += 32) {
        int j = base + lane;
        float dx = X[j*3]-cx, dy = X[j*3+1]-cy, dz = X[j*3+2]-cz;
        bool inr = (dx*dx + dy*dy + dz*dz) < (0.1f * 0.1f);
        unsigned m = (unsigned)__ballot(inr);
        int pos = count + __popc(m & ((1u << lane) - 1u));
        if (inr && pos < 32) sidx[pos] = j;
        count += __popc(m);
        if (count >= 32) break;
    }
    __syncthreads();
    int fill = (count > 0) ? sidx[0] : 0;
    __syncthreads();
    if (lane >= count) sidx[lane] = fill;
    __syncthreads();

    int j = sidx[lane];
    idx_out[(size_t)c * 32 + lane] = (float)j;
    size_t row = (size_t)c * 32 + lane;
    f16* dst = x0 + row * 32;
    dst[0] = (f16)(X[j*3+0] - cx);
    dst[1] = (f16)(X[j*3+1] - cy);
    dst[2] = (f16)(X[j*3+2] - cz);
    const float* P = points + ((size_t)b * Npts + j) * Cch;
    #pragma unroll
    for (int ch = 0; ch < 16; ++ch) dst[3 + ch] = (f16)P[ch];
    #pragma unroll
    for (int ch = 19; ch < 32; ++ch) dst[ch] = (f16)0.f;
}

// ---------------------------------------------------------------------------
// WMMA GEMM: C[M,128] = A[M,Kpad] @ W + bias (+res) (relu?), f16 in, f32 acc.
// Block = 256 threads = 8 waves; block -> 64-row super-tile, wave -> ntile.
// Each wave register-blocks MT=4 m-tiles: B fragment loaded once per K-step
// and reused by 4 back-to-back v_wmma_f32_16x16x32_f16.
// ---------------------------------------------------------------------------
__global__ __launch_bounds__(256) void gemm_wmma(const f16* __restrict__ A, int lda, int KC,
                                                 const f16* __restrict__ frag,
                                                 const float* __restrict__ bias,
                                                 f16* __restrict__ C,
                                                 const f16* __restrict__ res, int relu) {
    int lane  = threadIdx.x & 31;
    int ntile = threadIdx.x >> 5;
    int mtile0 = blockIdx.x * MT;
    int k0 = (lane & 16) ? 8 : 0;       // A layout: upper half-wave holds K+8
    v8f acc[MT];
    #pragma unroll
    for (int mt = 0; mt < MT; ++mt) acc[mt] = (v8f){0.f,0.f,0.f,0.f,0.f,0.f,0.f,0.f};

    const f16* arow0 = A + (size_t)(mtile0 * 16 + (lane & 15)) * lda + k0;
    const f16* bptr  = frag + ((size_t)ntile * KC * 32 + lane) * 16;
    for (int kc = 0; kc < KC; ++kc) {
        v16h bmat = *(const v16h*)(bptr + (size_t)kc * 512);
        #pragma unroll
        for (int mt = 0; mt < MT; ++mt) {
            const f16* arow = arow0 + (size_t)mt * 16 * lda + kc * 32;
            union { v16h v; v8h h[2]; } au;
            au.h[0] = *(const v8h*)(arow);        // K = kc*32 + k0 + 0..7
            au.h[1] = *(const v8h*)(arow + 16);   // K = kc*32 + k0 + 16..23
            acc[mt] = __builtin_amdgcn_wmma_f32_16x16x32_f16(
                false, au.v, false, bmat, (short)0, acc[mt], false, false);
        }
    }
    int n = ntile * 16 + (lane & 15);
    float bv = bias[n];
    #pragma unroll
    for (int mt = 0; mt < MT; ++mt) {
        int mbase = (mtile0 + mt) * 16 + ((lane & 16) >> 1);  // +8 for upper half-wave
        #pragma unroll
        for (int r = 0; r < 8; ++r) {
            int m = mbase + r;
            float v = acc[mt][r] + bv;
            if (res) v += (float)res[(size_t)m * DMODEL + n];
            if (relu) v = v > 0.f ? v : 0.f;
            C[(size_t)m * DMODEL + n] = (f16)v;
        }
    }
}

// ---------------------------------------------------------------------------
// Inner attention: one wave32 per (group, head). S=32, dk=dv=32.
// K/V tiles staged in LDS, softmax per query row (one lane per row).
// ---------------------------------------------------------------------------
__global__ __launch_bounds__(32) void inner_attn(const f16* __restrict__ Q,
                                                 const f16* __restrict__ K,
                                                 const f16* __restrict__ V,
                                                 f16* __restrict__ O) {
    __shared__ float sK[32 * 33];
    __shared__ float sV[32 * 33];
    int g = blockIdx.x >> 2;
    int h = blockIdx.x & 3;
    int lane = threadIdx.x;
    size_t rb = (size_t)g * 32;
    const f16* krow = K + (rb + lane) * DMODEL + h * 32;
    const f16* vrow = V + (rb + lane) * DMODEL + h * 32;
    #pragma unroll
    for (int d = 0; d < 32; ++d) { sK[lane*33 + d] = (float)krow[d]; sV[lane*33 + d] = (float)vrow[d]; }
    __syncthreads();
    const f16* qrow = Q + (rb + lane) * DMODEL + h * 32;
    float q[32];
    #pragma unroll
    for (int d = 0; d < 32; ++d) q[d] = (float)qrow[d];
    float sc[32]; float mx = -3.4e38f;
    for (int t = 0; t < 32; ++t) {
        float s = 0.f;
        #pragma unroll
        for (int d = 0; d < 32; ++d) s += q[d] * sK[t*33 + d];
        s *= RSCALE; sc[t] = s; mx = mx > s ? mx : s;
    }
    float den = 0.f;
    for (int t = 0; t < 32; ++t) { float e = __expf(sc[t] - mx); sc[t] = e; den += e; }
    float o[32];
    #pragma unroll
    for (int d = 0; d < 32; ++d) o[d] = 0.f;
    for (int t = 0; t < 32; ++t) {
        float w = sc[t];
        #pragma unroll
        for (int d = 0; d < 32; ++d) o[d] += w * sV[t*33 + d];
    }
    float rd = 1.f / den;
    f16* orow = O + (rb + lane) * DMODEL + h * 32;
    #pragma unroll
    for (int d = 0; d < 32; ++d) orow[d] = (f16)(o[d] * rd);
}

// ---------------------------------------------------------------------------
// Final attention: query = zeros -> Q = bias_q only. One wave per group:
// lane t scores K row t, wave softmax via shuffles, weighted V sum -> f32 out.
// ---------------------------------------------------------------------------
__global__ __launch_bounds__(32) void final_attn(const f16* __restrict__ K,
                                                 const f16* __restrict__ V,
                                                 const float* __restrict__ bq,
                                                 float* __restrict__ out) {
    int g = blockIdx.x;
    int lane = threadIdx.x;
    size_t rb = (size_t)g * 32;
    const f16* krow = K + (rb + lane) * DMODEL;
    float s = 0.f;
    for (int d = 0; d < DMODEL; ++d) s += bq[d] * (float)krow[d];
    s *= RSCALE;
    float m = s;
    for (int off = 16; off; off >>= 1) { float o2 = __shfl_xor(m, off, 32); m = m > o2 ? m : o2; }
    float e = __expf(s - m);
    float den = e;
    for (int off = 16; off; off >>= 1) den += __shfl_xor(den, off, 32);
    float w = e / den;
    float acc[4] = {0.f, 0.f, 0.f, 0.f};
    for (int t = 0; t < 32; ++t) {
        float wt = __shfl(w, t, 32);
        const f16* vrow = V + (rb + t) * DMODEL + lane;
        acc[0] += wt * (float)vrow[0];
        acc[1] += wt * (float)vrow[32];
        acc[2] += wt * (float)vrow[64];
        acc[3] += wt * (float)vrow[96];
    }
    #pragma unroll
    for (int c2 = 0; c2 < 4; ++c2) out[(size_t)g * DMODEL + lane + 32 * c2] = acc[c2];
}

// ---------------------------------------------------------------------------
extern "C" void kernel_launch(void* const* d_in, const int* in_sizes, int n_in,
                              void* d_out, int out_size, void* d_ws, size_t ws_size,
                              hipStream_t stream) {
    (void)in_sizes; (void)n_in; (void)out_size; (void)ws_size;
    const float* xyz    = (const float*)d_in[0];
    const float* points = (const float*)d_in[1];

    float* out        = (float*)d_out;
    float* out_newxyz = out;
    float* out_final  = out + Bdim * G * 3;
    float* out_idx    = out + Bdim * G * 3 + Bdim * G * DMODEL;

    char* ws = (char*)d_ws;
    size_t off = 0;
    auto take = [&](size_t bytes) -> char* {
        char* p = ws + off; off = (off + bytes + 255) & ~(size_t)255; return p;
    };
    float* nx_ws = (float*)take((size_t)Bdim * G * 3 * sizeof(float));
    f16*   frag  = (f16*)  take((size_t)18 * 16384 * sizeof(f16));
    f16*   x0    = (f16*)  take((size_t)MROWS * 32 * sizeof(f16));
    f16*   P[5];
    for (int i = 0; i < 5; ++i) P[i] = (f16*)take((size_t)MROWS * DMODEL * sizeof(f16));

    // --- weight fragment prep (18 GEMM weights; final q.W is unused) ---
    const int mats_w[18] = {2,4,6,8,10,12,14,16, 18,20,22,24,26,28,30,32, 36,38};
    const int mats_k[18] = {19,128,128,128,128,128,128,128,
                            128,128,128,128,128,128,128,128, 128,128};
    for (int i = 0; i < 18; ++i) {
        int K = mats_k[i]; int KC = (K + 31) / 32;
        int total = 8 * KC * 512;
        prep_frag<<<(total + 255) / 256, 256, 0, stream>>>(
            (const float*)d_in[mats_w[i]], K, KC, frag + (size_t)i * 16384);
    }

    // --- sampling + grouping ---
    fps_kernel<<<Bdim, 256, 0, stream>>>(xyz, nx_ws, out_newxyz);
    group_kernel<<<Bdim * G, 32, 0, stream>>>(xyz, points, nx_ws, x0, out_idx);

    auto bias = [&](int i) { return (const float*)d_in[i]; };
    auto gemm = [&](const f16* A, int lda, int KC, int slot, const float* bs,
                    f16* C, const f16* res, int relu) {
        gemm_wmma<<<MROWS / (16 * MT), 256, 0, stream>>>(A, lda, KC,
                                                         frag + (size_t)slot * 16384,
                                                         bs, C, res, relu);
    };

    // --- two transformer blocks ---
    const f16* in = x0; int lda = 32, KC = 1;
    for (int blk = 0; blk < 2; ++blk) {
        f16* F0 = P[0]; f16* F1 = P[1]; f16* F2 = P[2];
        f16* F3 = (blk == 0) ? P[3] : P[4];   // blk1 input lives in P[3]
        int s  = blk * 8;
        int bb = 2 + blk * 16;
        gemm(in, lda, KC, s + 0, bias(bb + 1),  F0, nullptr, 1);  // t1 = relu(x@Wp0+b)
        gemm(F0, 128, 4, s + 1, bias(bb + 3),  F1, nullptr, 0);   // x2 = t1@Wp1+b
        gemm(F1, 128, 4, s + 2, bias(bb + 5),  F0, nullptr, 0);   // q
        gemm(F1, 128, 4, s + 3, bias(bb + 7),  F2, nullptr, 0);   // k
        gemm(F1, 128, 4, s + 4, bias(bb + 9),  F3, nullptr, 0);   // v
        inner_attn<<<Bdim * G * NH, 32, 0, stream>>>(F0, F2, F3, F1);  // att -> F1
        gemm(F1, 128, 4, s + 5, bias(bb + 11), F0, nullptr, 0);   // x3 = att@Wo+b
        gemm(F0, 128, 4, s + 6, bias(bb + 13), F2, nullptr, 1);   // t2 = relu(x3@Wf0+b)
        gemm(F2, 128, 4, s + 7, bias(bb + 15), F3, F0, 0);        // out = t2@Wf1+b + x3
        in = F3; lda = 128; KC = 4;
    }

    // --- final attention (Q = bias only, since query = zeros) ---
    gemm(in, 128, 4, 16, bias(37), P[0], nullptr, 0);  // K_f
    gemm(in, 128, 4, 17, bias(39), P[1], nullptr, 0);  // V_f
    final_attn<<<Bdim * G, 32, 0, stream>>>(P[0], P[1], (const float*)d_in[35], out_final);
}
